// GraphTransform_44590350467891
// MI455X (gfx1250) — matrix operand order repaired and checked by reading, not compiled, not run on hardware
//
#include <hip/hip_runtime.h>

#define NN 10000
#define EE 100000
#define DD 768
#define HH 1536

typedef __attribute__((ext_vector_type(16))) __bf16 v16bf;
typedef __attribute__((ext_vector_type(8)))  float  v8f;

// ---------------- CDNA5 async global->LDS copy (ASYNCcnt path) ----------------
// ISA 15.18.3 opcode 98: GLOBAL_LOAD_ASYNC_TO_LDS_B128, "GV" mode:
//   vdst = LDS byte offset VGPR, vaddr = 64-bit global address VGPR pair.
__device__ __forceinline__ void async_copy_b128(const void* gptr, void* lptr) {
    unsigned lds_off = (unsigned)(unsigned long long)(uintptr_t)lptr; // low 32b of generic LDS ptr = LDS offset
    unsigned long long ga = (unsigned long long)(uintptr_t)gptr;
    asm volatile("global_load_async_to_lds_b128 %0, %1, off"
                 :: "v"(lds_off), "v"(ga)
                 : "memory");
}

// ---------------- float atomic max via integer punning ----------------
__device__ __forceinline__ void atomicMaxF(float* a, float v) {
    if (v >= 0.0f) atomicMax((int*)a, __float_as_int(v));
    else           atomicMin((unsigned int*)a, __float_as_uint(v));
}

// ---------------- init accumulators ----------------
__global__ void k_init(float* __restrict__ mx, float* __restrict__ sm,
                       float* __restrict__ ag, int n) {
    int i = blockIdx.x * blockDim.x + threadIdx.x;
    if (i < n) {
        mx[i] = __int_as_float(0xff800000); // -inf
        sm[i] = 0.0f;
        ag[i] = 0.0f;
    }
}

// ---------------- edge pass 1: segment max of logits ----------------
__global__ __launch_bounds__(192) void k_edge_max(const float* __restrict__ x,
        const int* __restrict__ ei, const float* __restrict__ tp,
        float* __restrict__ mx) {
    const int e   = blockIdx.x;
    const int src = ei[e];
    const int dst = ei[EE + e];
    const float t = tp[0];
    const int c   = threadIdx.x * 4;
    float4 v = *(const float4*)(x + (size_t)src * DD + c);
    float* mrow = mx + (size_t)dst * DD + c;
    atomicMaxF(mrow + 0, (fmaxf(v.x, 0.f) + 1e-7f) * t);
    atomicMaxF(mrow + 1, (fmaxf(v.y, 0.f) + 1e-7f) * t);
    atomicMaxF(mrow + 2, (fmaxf(v.z, 0.f) + 1e-7f) * t);
    atomicMaxF(mrow + 3, (fmaxf(v.w, 0.f) + 1e-7f) * t);
}

// ---------------- edge pass 2: exp-sum ----------------
__global__ __launch_bounds__(192) void k_edge_expsum(const float* __restrict__ x,
        const int* __restrict__ ei, const float* __restrict__ tp,
        const float* __restrict__ mx, float* __restrict__ sm) {
    const int e   = blockIdx.x;
    const int src = ei[e];
    const int dst = ei[EE + e];
    const float t = tp[0];
    const int c   = threadIdx.x * 4;
    float4 v = *(const float4*)(x  + (size_t)src * DD + c);
    float4 m = *(const float4*)(mx + (size_t)dst * DD + c);
    float* srow = sm + (size_t)dst * DD + c;
    atomicAdd(srow + 0, __expf((fmaxf(v.x, 0.f) + 1e-7f) * t - m.x));
    atomicAdd(srow + 1, __expf((fmaxf(v.y, 0.f) + 1e-7f) * t - m.y));
    atomicAdd(srow + 2, __expf((fmaxf(v.z, 0.f) + 1e-7f) * t - m.z));
    atomicAdd(srow + 3, __expf((fmaxf(v.w, 0.f) + 1e-7f) * t - m.w));
}

// ---------------- edge pass 3: weighted aggregate ----------------
__global__ __launch_bounds__(192) void k_edge_agg(const float* __restrict__ x,
        const int* __restrict__ ei, const float* __restrict__ tp,
        const float* __restrict__ mx, const float* __restrict__ sm,
        float* __restrict__ ag) {
    const int e   = blockIdx.x;
    const int src = ei[e];
    const int dst = ei[EE + e];
    const float t = tp[0];
    const int c   = threadIdx.x * 4;
    float4 v = *(const float4*)(x  + (size_t)src * DD + c);
    float4 m = *(const float4*)(mx + (size_t)dst * DD + c);
    float4 s = *(const float4*)(sm + (size_t)dst * DD + c);
    float* arow = ag + (size_t)dst * DD + c;
    float g0 = fmaxf(v.x, 0.f) + 1e-7f;
    float g1 = fmaxf(v.y, 0.f) + 1e-7f;
    float g2 = fmaxf(v.z, 0.f) + 1e-7f;
    float g3 = fmaxf(v.w, 0.f) + 1e-7f;
    atomicAdd(arow + 0, g0 * __expf(g0 * t - m.x) / (s.x + 1e-16f));
    atomicAdd(arow + 1, g1 * __expf(g1 * t - m.y) / (s.y + 1e-16f));
    atomicAdd(arow + 2, g2 * __expf(g2 * t - m.z) / (s.z + 1e-16f));
    atomicAdd(arow + 3, g3 * __expf(g3 * t - m.w) / (s.w + 1e-16f));
}

// ---------------- prep: A1 = bf16(agg + x) ----------------
__global__ void k_prep_a(const float* __restrict__ ag, const float* __restrict__ x,
                         __bf16* __restrict__ a1, int n) {
    int i = blockIdx.x * blockDim.x + threadIdx.x;
    if (i < n) a1[i] = (__bf16)(ag[i] + x[i]);
}

// ---------------- float -> bf16 copy (weights) ----------------
__global__ void k_cvt(const float* __restrict__ in, __bf16* __restrict__ out, int n) {
    int i = blockIdx.x * blockDim.x + threadIdx.x;
    if (i < n) out[i] = (__bf16)in[i];
}

// ---------------- fold BN into scale/shift ----------------
__global__ void k_bnprep(const float* __restrict__ b1, const float* __restrict__ g,
                         const float* __restrict__ be, const float* __restrict__ mn,
                         const float* __restrict__ vr, float* __restrict__ sc,
                         float* __restrict__ sh) {
    int i = blockIdx.x * blockDim.x + threadIdx.x;
    if (i < HH) {
        float s = g[i] * rsqrtf(vr[i] + 1e-5f);
        sc[i] = s;
        sh[i] = (b1[i] - mn[i]) * s + be[i];
    }
}

// ---------------- bf16 WMMA GEMM, 64x256 block tile, 8 waves ----------------
// Double-buffered async global->LDS staging (5 async b128 per wave per tile).
// EPI==1: z = relu(acc*scale[n] + shift[n]) -> bf16 out
// EPI==2: out = x_res + acc + bias[n]       -> fp32 out
template <int EPI>
__global__ __launch_bounds__(256) void k_gemm(const __bf16* __restrict__ A,
        const __bf16* __restrict__ B, int M, int K, int Ntot,
        const float* __restrict__ p0, const float* __restrict__ p1,
        const float* __restrict__ xres, __bf16* __restrict__ obf,
        float* __restrict__ of) {
    __shared__ __bf16 As[2][64][40];   // 64x32 bf16 tile, padded rows (80B, 16B aligned)
    __shared__ __bf16 Bs[2][32][264];  // 32x256 bf16 tile, padded rows (528B, 16B aligned)

    const int tid  = threadIdx.x;
    const int lane = tid & 31;
    const int w    = tid >> 5;
    const int wm   = w >> 2;   // 0..1 (32 rows each)
    const int wn   = w & 3;    // 0..3 (64 cols each)
    const int mbase = blockIdx.x * 64;
    const int nbase = blockIdx.y * 256;

    v8f acc[2][4];
#pragma unroll
    for (int i = 0; i < 2; i++)
#pragma unroll
        for (int j = 0; j < 4; j++) acc[i][j] = v8f{};

    // staging assignment: 1 A chunk + 4 B chunks of 16B per thread per K-step
    const int ar = tid >> 2;                                   // 0..63
    const int ac = (tid & 3) * 8;                              // 0,8,16,24
    const int arow = (mbase + ar < M) ? (mbase + ar) : (M - 1); // clamp: rows independent in WMMA

    auto issue_tile = [&](int buf, int kk) {
        async_copy_b128(A + (size_t)arow * K + kk + ac, &As[buf][ar][ac]);
#pragma unroll
        for (int j = 0; j < 4; j++) {
            int c  = tid + 256 * j;
            int br = c >> 5;            // 0..31
            int bc = (c & 31) * 8;      // 0..248
            async_copy_b128(B + (size_t)(kk + br) * Ntot + nbase + bc, &Bs[buf][br][bc]);
        }
    };

    const int nsteps = K >> 5;
    issue_tile(0, 0);

    for (int step = 0; step < nsteps; ++step) {
        const int cur = step & 1;
        if (step + 1 < nsteps) {
            issue_tile(cur ^ 1, (step + 1) << 5);
            asm volatile("s_wait_asynccnt 0x5" ::: "memory"); // tile `cur` landed
        } else {
            asm volatile("s_wait_asynccnt 0x0" ::: "memory");
        }
        __syncthreads();

        // ---- fragments per documented CDNA5 bf16 layouts ----
        v16bf af[2];
#pragma unroll
        for (int mi = 0; mi < 2; mi++) {
            int m  = wm * 32 + mi * 16 + (lane & 15);
            int kb = (lane >> 4) * 8;   // lanes 0-15: K0-7/16-23; lanes 16-31: K8-15/24-31
            union { v16bf v; uint4 q[2]; } u;
            u.q[0] = *(const uint4*)(&As[cur][m][kb]);
            u.q[1] = *(const uint4*)(&As[cur][m][kb + 16]);
            af[mi] = u.v;
        }
        v16bf bfr[4];
#pragma unroll
        for (int ni = 0; ni < 4; ni++) {
            int n = wn * 64 + ni * 16 + (lane & 15);  // N = lane%16; K per element matches A (table order)
            int kb = (lane >> 4) * 8;
            v16bf t;
#pragma unroll
            for (int s = 0; s < 8; s++) {
                t[s]     = Bs[cur][kb + s][n];
                t[s + 8] = Bs[cur][kb + 16 + s][n];
            }
            bfr[ni] = t;
        }

#pragma unroll
        for (int mi = 0; mi < 2; mi++)
#pragma unroll
            for (int ni = 0; ni < 4; ni++)
                acc[mi][ni] = __builtin_amdgcn_wmma_f32_16x16x32_bf16(
                    false, af[mi], false, bfr[ni], (short)0, acc[mi][ni],
                    false, false);

        __syncthreads(); // protect buffer `cur` from being overwritten at step+2's issue
    }

    // ---- epilogue ----
    const int ml = lane >> 4;   // 0/1 -> +8 rows
    const int nl = lane & 15;
#pragma unroll
    for (int ni = 0; ni < 4; ni++) {
        int gn = nbase + wn * 64 + ni * 16 + nl;
        float s0 = p0[gn];
        float s1 = (EPI == 1) ? p1[gn] : 0.0f;
#pragma unroll
        for (int mi = 0; mi < 2; mi++) {
#pragma unroll
            for (int v = 0; v < 8; v++) {
                int gm = mbase + wm * 32 + mi * 16 + v + 8 * ml;
                if (gm < M) {
                    float a = acc[mi][ni][v];
                    if (EPI == 1) {
                        float z = a * s0 + s1;
                        z = z > 0.0f ? z : 0.0f;
                        obf[(size_t)gm * Ntot + gn] = (__bf16)z;
                    } else {
                        of[(size_t)gm * Ntot + gn] =
                            xres[(size_t)gm * Ntot + gn] + a + s0;
                    }
                }
            }
        }
    }
}

extern "C" void kernel_launch(void* const* d_in, const int* in_sizes, int n_in,
                              void* d_out, int out_size, void* d_ws, size_t ws_size,
                              hipStream_t stream) {
    const float* x     = (const float*)d_in[0];
    const int*   ei    = (const int*)d_in[1];
    const float* t     = (const float*)d_in[3];
    const float* W1    = (const float*)d_in[4];
    const float* b1    = (const float*)d_in[5];
    const float* gamma = (const float*)d_in[6];
    const float* beta  = (const float*)d_in[7];
    const float* mean  = (const float*)d_in[8];
    const float* var   = (const float*)d_in[9];
    const float* W2    = (const float*)d_in[10];
    const float* b2    = (const float*)d_in[11];
    float* out = (float*)d_out;

    char* ws = (char*)d_ws;
    size_t off = 0;
    auto alloc = [&](size_t bytes) {
        void* p = ws + off;
        off = (off + bytes + 255) & ~(size_t)255;
        return p;
    };
    float*  maxbuf = (float*)alloc((size_t)NN * DD * 4);
    float*  sumbuf = (float*)alloc((size_t)NN * DD * 4);
    float*  agg    = (float*)alloc((size_t)NN * DD * 4);
    __bf16* a1     = (__bf16*)alloc((size_t)NN * DD * 2);
    __bf16* w1b    = (__bf16*)alloc((size_t)DD * HH * 2);
    __bf16* w2b    = (__bf16*)alloc((size_t)HH * DD * 2);
    float*  bnsc   = (float*)alloc((size_t)HH * 4);
    float*  bnsh   = (float*)alloc((size_t)HH * 4);
    // z1 (bf16 [N,H], 30.72MB) aliases maxbuf (fp32 [N,D], 30.72MB): maxbuf is
    // dead after k_edge_agg, z1 is first written in GEMM1 afterwards.
    __bf16* z1 = (__bf16*)maxbuf;

    const int ND = NN * DD;
    k_init<<<(ND + 255) / 256, 256, 0, stream>>>(maxbuf, sumbuf, agg, ND);
    k_edge_max<<<EE, 192, 0, stream>>>(x, ei, t, maxbuf);
    k_edge_expsum<<<EE, 192, 0, stream>>>(x, ei, t, maxbuf, sumbuf);
    k_edge_agg<<<EE, 192, 0, stream>>>(x, ei, t, maxbuf, sumbuf, agg);
    k_prep_a<<<(ND + 255) / 256, 256, 0, stream>>>(agg, x, a1, ND);
    k_cvt<<<(DD * HH + 255) / 256, 256, 0, stream>>>(W1, w1b, DD * HH);
    k_cvt<<<(HH * DD + 255) / 256, 256, 0, stream>>>(W2, w2b, HH * DD);
    k_bnprep<<<(HH + 255) / 256, 256, 0, stream>>>(b1, gamma, beta, mean, var, bnsc, bnsh);

    dim3 g1((NN + 63) / 64, HH / 256);
    k_gemm<1><<<g1, 256, 0, stream>>>(a1, w1b, NN, DD, HH, bnsc, bnsh,
                                      nullptr, z1, nullptr);
    dim3 g2((NN + 63) / 64, DD / 256);
    k_gemm<2><<<g2, 256, 0, stream>>>(z1, w2b, NN, HH, DD, b2, nullptr,
                                      x, nullptr, out);
}
